// TrajectoryGenerator_85375359910301
// MI455X (gfx1250) — matrix operand order repaired and checked by reading, not compiled, not run on hardware
//
#include <hip/hip_runtime.h>
#include <math.h>

// ---------------------------------------------------------------------------
// TrajectoryGenerator forward for MI455X (gfx1250, wave32, WMMA).
//
// Pooling-net layer 1 is affine in [se(p_i-p_j)|h_j|h_i], so
//   y1[i,j] = relu(base_i + basej[j]),
//   base_i  = m1_b + M1r@se_b + h_i@M1i.T + p_i@(M1r@se_W).T
//   basej_j =                    h_j@M1j.T - p_j@(M1r@se_W).T
// Only the post-ReLU 64->16 layer is pairwise -> WMMA hot loop
// (v_wmma_f32_16x16x32_f16, 2 per 16-j tile). basej is built with WMMA in
// transposed orientation (weights as A, [h_j|p_j|pad] as B) so C-frag stores
// are contiguous b128 LDS stores. All fragment builds are branchless
// (cndmask selects, contiguous vector loads) to avoid exec-masked scalar
// load serialization. Main-loop A operands use packed-f16 add/max.
// ---------------------------------------------------------------------------

#define OBS_LEN 8
#define PRED_LEN 12
#define NAGT 512
#define HDIM 16
#define NEGV (-1e30f)
#define LOG2PI 1.8378770664093453f

typedef __attribute__((ext_vector_type(16))) _Float16 v16h;
typedef __attribute__((ext_vector_type(8)))  _Float16 v8h;
typedef __attribute__((ext_vector_type(8)))  float    v8f;

union FragA { v16h v; _Float16 h[16]; };
union FragC { v8f  v; float    f[8];  };
union F8h   { v8h  v; _Float16 h[8];  };

__device__ __forceinline__ float sigf(float x) { return 1.0f / (1.0f + __expf(-x)); }

__device__ __forceinline__ v8h relu8h(v8h x) {
  v8h z = {};
#if __has_builtin(__builtin_elementwise_max)
  return __builtin_elementwise_max(x, z);
#else
  F8h a; a.v = x;
  #pragma unroll
  for (int e = 0; e < 8; e++) a.h[e] = a.h[e] > (_Float16)0 ? a.h[e] : (_Float16)0;
  return a.v;
#endif
}

// ---------------------------------------------------------------------------
// Kernel 1: final-goal embedding + 8-step encoder LSTM + carry init.
// ---------------------------------------------------------------------------
__global__ void init_encoder_kernel(
    const float* __restrict__ traj_rel, const float* __restrict__ obs_pos,
    const float* __restrict__ goal,
    const float* __restrict__ h0, const float* __restrict__ c0,
    const float* __restrict__ enc_W, const float* __restrict__ enc_b,
    const float* __restrict__ goal_W, const float* __restrict__ goal_b,
    const float* __restrict__ Wih, const float* __restrict__ Whh,
    const float* __restrict__ bih, const float* __restrict__ bhh,
    float* __restrict__ wh, float* __restrict__ wc, float* __restrict__ wctx,
    float* __restrict__ wout, float* __restrict__ wcurr, float* __restrict__ wfg,
    float* __restrict__ wnll)
{
  __shared__ float sWih[64 * 16], sWhh[64 * 16], sb[64];
  __shared__ float sEW[32], sEb[16], sGW[32], sGb[16];
  const int tid = threadIdx.x;
  for (int e = tid; e < 64 * 16; e += 512) { sWih[e] = Wih[e]; sWhh[e] = Whh[e]; }
  if (tid < 64) sb[tid] = bih[tid] + bhh[tid];
  if (tid < 32) { sEW[tid] = enc_W[tid]; sGW[tid] = goal_W[tid]; }
  if (tid < 16) { sEb[tid] = enc_b[tid]; sGb[tid] = goal_b[tid]; }
  __syncthreads();

  const int n = tid;
  float h[16], c[16];
  #pragma unroll
  for (int k = 0; k < 16; k++) { h[k] = h0[n * 16 + k]; c[k] = c0[n * 16 + k]; }

  for (int t = 0; t < OBS_LEN; t++) {
    const float x0 = traj_rel[(t * NAGT + n) * 2 + 0];
    const float x1 = traj_rel[(t * NAGT + n) * 2 + 1];
    float emb[16];
    #pragma unroll
    for (int e = 0; e < 16; e++)
      emb[e] = fmaxf(0.f, sEW[e * 2] * x0 + sEW[e * 2 + 1] * x1 + sEb[e]);
    float g[64];
    for (int r = 0; r < 64; r++) {
      float a = sb[r];
      #pragma unroll
      for (int k = 0; k < 16; k++) a += sWih[r * 16 + k] * emb[k] + sWhh[r * 16 + k] * h[k];
      g[r] = a;
    }
    #pragma unroll
    for (int k = 0; k < 16; k++) {
      const float ig = sigf(g[k]), fg = sigf(g[16 + k]);
      const float gg = tanhf(g[32 + k]), og = sigf(g[48 + k]);
      const float c2 = fg * c[k] + ig * gg;
      c[k] = c2;
      h[k] = og * tanhf(c2);
    }
  }
  #pragma unroll
  for (int k = 0; k < 16; k++) { wh[n * 16 + k] = h[k]; wc[n * 16 + k] = c[k]; wctx[n * 16 + k] = 0.f; }

  const float lx = obs_pos[((OBS_LEN - 1) * NAGT + n) * 2 + 0];
  const float ly = obs_pos[((OBS_LEN - 1) * NAGT + n) * 2 + 1];
  const float dx = goal[n * 2 + 0] - lx, dy = goal[n * 2 + 1] - ly;
  #pragma unroll
  for (int k = 0; k < 16; k++) wfg[n * 16 + k] = sGW[k * 2] * dx + sGW[k * 2 + 1] * dy + sGb[k];

  wout[n * 2 + 0] = traj_rel[((OBS_LEN - 1) * NAGT + n) * 2 + 0];
  wout[n * 2 + 1] = traj_rel[((OBS_LEN - 1) * NAGT + n) * 2 + 1];
  wcurr[n * 2 + 0] = lx;
  wcurr[n * 2 + 1] = ly;
  if (n == 0) wnll[0] = 0.f;
}

// ---------------------------------------------------------------------------
// Kernel 2 (per step): goal-distance embedding, decoder input, pred LSTM.
// ---------------------------------------------------------------------------
__global__ void dec_pre_kernel(
    const float* __restrict__ goal,
    const float* __restrict__ gd_W, const float* __restrict__ gd_b,
    const float* __restrict__ dec_W, const float* __restrict__ dec_b,
    const float* __restrict__ Wih, const float* __restrict__ Whh,
    const float* __restrict__ bih, const float* __restrict__ bhh,
    const float* __restrict__ wctx, const float* __restrict__ wout,
    const float* __restrict__ wcurr,
    float* __restrict__ wh, float* __restrict__ wc, float* __restrict__ wedg)
{
  __shared__ float sWih[64 * 16], sWhh[64 * 16], sb[64];
  __shared__ float sDW[16 * 34], sDb[16], sGW[32], sGb[16];
  const int tid = threadIdx.x;
  for (int e = tid; e < 64 * 16; e += 512) { sWih[e] = Wih[e]; sWhh[e] = Whh[e]; }
  for (int e = tid; e < 16 * 34; e += 512) sDW[e] = dec_W[e];
  if (tid < 64) sb[tid] = bih[tid] + bhh[tid];
  if (tid < 32) sGW[tid] = gd_W[tid];
  if (tid < 16) { sGb[tid] = gd_b[tid]; sDb[tid] = dec_b[tid]; }
  __syncthreads();

  const int n = tid;
  const float dx = goal[n * 2 + 0] - wcurr[n * 2 + 0];
  const float dy = goal[n * 2 + 1] - wcurr[n * 2 + 1];
  float inp[34];
  #pragma unroll
  for (int k = 0; k < 16; k++) inp[k] = wctx[n * 16 + k];
  inp[16] = wout[n * 2 + 0];
  inp[17] = wout[n * 2 + 1];
  #pragma unroll
  for (int k = 0; k < 16; k++) {
    const float e = sGW[k * 2] * dx + sGW[k * 2 + 1] * dy + sGb[k];
    inp[18 + k] = e;
    wedg[n * 16 + k] = e;
  }
  float emb[16];
  for (int e = 0; e < 16; e++) {
    float a = sDb[e];
    #pragma unroll
    for (int k = 0; k < 34; k++) a += sDW[e * 34 + k] * inp[k];
    emb[e] = fmaxf(0.f, a);
  }
  float h[16], c[16];
  #pragma unroll
  for (int k = 0; k < 16; k++) { h[k] = wh[n * 16 + k]; c[k] = wc[n * 16 + k]; }
  float g[64];
  for (int r = 0; r < 64; r++) {
    float a = sb[r];
    #pragma unroll
    for (int k = 0; k < 16; k++) a += sWih[r * 16 + k] * emb[k] + sWhh[r * 16 + k] * h[k];
    g[r] = a;
  }
  #pragma unroll
  for (int k = 0; k < 16; k++) {
    const float ig = sigf(g[k]), fg = sigf(g[16 + k]);
    const float gg = tanhf(g[32 + k]), og = sigf(g[48 + k]);
    const float c2 = fg * c[k] + ig * gg;
    wc[n * 16 + k] = c2;
    wh[n * 16 + k] = og * tanhf(c2);
  }
}

// ---------------------------------------------------------------------------
// Kernel 3: factored WMMA pooling net.
// 64 blocks x 256 threads (8 waves); wave w owns i = blockIdx*8 + w.
// ---------------------------------------------------------------------------
__global__ void __launch_bounds__(256) pool_kernel(
    const float* __restrict__ pos, const float* __restrict__ wh,
    const int* __restrict__ nei,                 // pre-offset to step t: (N,N)
    const float* __restrict__ se_W, const float* __restrict__ se_b,
    const float* __restrict__ m1_W, const float* __restrict__ m1_b,
    const float* __restrict__ m2_W, const float* __restrict__ m2_b,
    float* __restrict__ ctx_out)
{
  __shared__ _Float16 sbj[256 * 64];           // 32 KB: per-pass j-basis (row-major)
  __shared__ _Float16 sbasei[8 * 64];          //  1 KB: per-wave i-basis (f16)
  __shared__ float    sW2[64 * 2];             // folded pos weights M1r@se_W
  __shared__ float    sbp[64];                 // folded bias m1_b + M1r@se_b
  __shared__ unsigned char sh_mask[8 * NAGT];  //  4 KB

  const int tid = threadIdx.x, lane = tid & 31, wv = tid >> 5;
  const int i = blockIdx.x * 8 + wv;
  const int lh = lane >> 4, mrow = lane & 15;

  // ---- Phase 0: fold se layer into m1; stage neighbor mask -------------
  if (tid < 64) {
    const int n = tid;
    float w0 = 0.f, w1 = 0.f, bb = m1_b[n];
    for (int k = 0; k < 32; k++) {
      const float m = m1_W[n * 64 + k];
      w0 += m * se_W[k * 2 + 0];
      w1 += m * se_W[k * 2 + 1];
      bb += m * se_b[k];
    }
    sW2[n * 2 + 0] = w0;
    sW2[n * 2 + 1] = w1;
    sbp[n] = bb;
  }
  {
    const int* row = nei + i * NAGT;
    __builtin_prefetch(row, 0, 1);             // global_prefetch_b8
    #pragma unroll
    for (int q = 0; q < 16; q++) {
      const int j = lane * 16 + q;
      sh_mask[wv * NAGT + j] = (unsigned char)(row[j] > 0);
    }
  }
  __syncthreads();

  // ---- Phase 0.5: per-wave i-basis (f16); persistent fragments ---------
  {
    const float px = pos[i * 2 + 0], py = pos[i * 2 + 1];
    #pragma unroll
    for (int q = 0; q < 2; q++) {
      const int n = lane * 2 + q;
      float a = sbp[n] + px * sW2[n * 2 + 0] + py * sW2[n * 2 + 1];
      for (int k = 0; k < 16; k++) a += wh[i * 16 + k] * m1_W[n * 64 + 48 + k];
      sbasei[wv * 64 + n] = (_Float16)a;
    }
  }
  // GEMM2 B fragments: B[k][n] = m2_W[n][k] (contiguous in e -> vector loads)
  FragA b2[2];
  #pragma unroll
  for (int s = 0; s < 2; s++)
    #pragma unroll
    for (int e = 0; e < 16; e++)
      b2[s].h[e] = (_Float16)m2_W[mrow * 64 + s * 32 + lh * 8 + (e & 7) + (e >> 3) * 16];
  const float bias2 = m2_b[mrow];
  // basej-GEMM A fragments (weights as A, transposed orientation).
  // Fragment algebra: e<8 -> k = lh*8+e < 16 always (plain contiguous load);
  // e>=8 -> k >= 16 always (value is -W2 only for lh==0 && e in {8,9}).
  FragA aw[4];
  #pragma unroll
  for (int nt = 0; nt < 4; nt++) {
    const int n = nt * 16 + mrow;
    #pragma unroll
    for (int e = 0; e < 8; e++)
      aw[nt].h[e] = (_Float16)m1_W[n * 64 + 32 + lh * 8 + e];   // M1j columns
    const float w20 = (lh == 0) ? -sW2[n * 2 + 0] : 0.f;        // cndmask
    const float w21 = (lh == 0) ? -sW2[n * 2 + 1] : 0.f;
    aw[nt].h[8] = (_Float16)w20;
    aw[nt].h[9] = (_Float16)w21;
    #pragma unroll
    for (int e = 10; e < 16; e++) aw[nt].h[e] = (_Float16)0.f;
  }
  __syncthreads();

  // Hoisted i-basis vectors (loop-invariant A-operand halves)
  const _Float16* sbi = &sbasei[wv * 64];
  const v8h bi0 = *(const v8h*)(sbi + lh * 8);
  const v8h bi1 = *(const v8h*)(sbi + 16 + lh * 8);
  const v8h bi2 = *(const v8h*)(sbi + 32 + lh * 8);
  const v8h bi3 = *(const v8h*)(sbi + 48 + lh * 8);

  FragC runmax;
  #pragma unroll
  for (int e = 0; e < 8; e++) runmax.f[e] = NEGV;

  for (int p = 0; p < 2; p++) {
    // ---- Phase 1: build basej tile (transposed WMMA -> b128 stores) ----
    #pragma unroll
    for (int r = 0; r < 2; r++) {
      const int jt = wv * 2 + r;               // local 16-j tile index
      const int j = p * 256 + jt * 16 + mrow;  // this lane's agent column
      // Branchless B operand: ALL lanes vector-load the h row (contiguous,
      // uniform control flow); e<8 -> h[lh*8+e], e>=8 -> [p_j | 0...] (lh==0) or 0.
      const float4 h0v = *(const float4*)(wh + j * 16 + 0);
      const float4 h1v = *(const float4*)(wh + j * 16 + 4);
      const float4 h2v = *(const float4*)(wh + j * 16 + 8);
      const float4 h3v = *(const float4*)(wh + j * 16 + 12);
      const float px = pos[j * 2 + 0], py = pos[j * 2 + 1];
      const float hv[16] = { h0v.x, h0v.y, h0v.z, h0v.w,
                             h1v.x, h1v.y, h1v.z, h1v.w,
                             h2v.x, h2v.y, h2v.z, h2v.w,
                             h3v.x, h3v.y, h3v.z, h3v.w };
      FragA bj;
      #pragma unroll
      for (int e = 0; e < 16; e++) {
        const float alt = (e == 8) ? px : (e == 9) ? py : 0.f;  // compile-time
        const float v = (e < 8) ? ((lh == 0) ? hv[e] : hv[(e + 8) & 15]) : ((lh == 0) ? alt : 0.f);  // cndmask
        bj.h[e] = (_Float16)v;
      }
      #pragma unroll
      for (int nt = 0; nt < 4; nt++) {
        FragC acc;
        #pragma unroll
        for (int e = 0; e < 8; e++) acc.f[e] = 0.f;
        acc.v = __builtin_amdgcn_wmma_f32_16x16x32_f16(false, aw[nt].v, false, bj.v,
                                                       (short)0, acc.v, false, false);
        F8h o;
        #pragma unroll
        for (int e = 0; e < 8; e++) o.h[e] = (_Float16)acc.f[e];
        // C elem v -> n = nt*16 + lh*8 + v (contiguous), col j fixed per lane
        *(v8h*)&sbj[(jt * 16 + mrow) * 64 + nt * 16 + lh * 8] = o.v;
      }
    }
    __syncthreads();

    // ---- Main loop: 16 j-tiles, A = relu(base_i + basej), 64->16 -------
    for (int jc = 0; jc < 16; jc++) {
      const _Float16* rj = &sbj[(jc * 16 + mrow) * 64];
      const v8h x0 = relu8h(*(const v8h*)(rj + lh * 8)      + bi0);
      const v8h x1 = relu8h(*(const v8h*)(rj + 16 + lh * 8) + bi1);
      const v8h x2 = relu8h(*(const v8h*)(rj + 32 + lh * 8) + bi2);
      const v8h x3 = relu8h(*(const v8h*)(rj + 48 + lh * 8) + bi3);
      FragA a0, a1;
      a0.v = __builtin_shufflevector(x0, x1, 0, 1, 2, 3, 4, 5, 6, 7,
                                     8, 9, 10, 11, 12, 13, 14, 15);
      a1.v = __builtin_shufflevector(x2, x3, 0, 1, 2, 3, 4, 5, 6, 7,
                                     8, 9, 10, 11, 12, 13, 14, 15);
      FragC acc2;
      #pragma unroll
      for (int e = 0; e < 8; e++) acc2.f[e] = bias2;
      acc2.v = __builtin_amdgcn_wmma_f32_16x16x32_f16(false, a0.v, false, b2[0].v,
                                                      (short)0, acc2.v, false, false);
      acc2.v = __builtin_amdgcn_wmma_f32_16x16x32_f16(false, a1.v, false, b2[1].v,
                                                      (short)0, acc2.v, false, false);
      #pragma unroll
      for (int e = 0; e < 8; e++) {
        const int jr = p * 256 + jc * 16 + e + lh * 8;
        const float y = fmaxf(0.f, acc2.f[e]);
        const float msk = sh_mask[wv * NAGT + jr] ? y : NEGV;
        runmax.f[e] = fmaxf(runmax.f[e], msk);
      }
    }
    __syncthreads();                           // sbj reused next pass
  }

  float mx = runmax.f[0];
  #pragma unroll
  for (int e = 1; e < 8; e++) mx = fmaxf(mx, runmax.f[e]);
  mx = fmaxf(mx, __shfl_xor(mx, 16, 32));      // combine half-lanes (same col)
  if (lane < 16) {
    ctx_out[i * HDIM + lane] = (mx <= NEGV * 0.5f) ? 0.f : mx;
  }
}

// ---------------------------------------------------------------------------
// Kernel 4 (per step): output head, NLL accumulation, sampled position.
// ---------------------------------------------------------------------------
__global__ void dec_head_kernel(
    const float* __restrict__ wctx, const float* __restrict__ wh,
    const float* __restrict__ wfg, const float* __restrict__ wedg,
    const float* __restrict__ h2p_W, const float* __restrict__ h2p_b,
    const float* __restrict__ target,
    const float* __restrict__ wcurr,
    float* __restrict__ wmu, float* __restrict__ wcurrs,
    float* __restrict__ wnll, float* __restrict__ out_sampled)
{
  __shared__ float sW[64], sb4[4], red[512];
  const int tid = threadIdx.x;
  if (tid < 64) sW[tid] = h2p_W[tid];
  if (tid < 4) sb4[tid] = h2p_b[tid];
  __syncthreads();

  const int n = tid;
  float cat[16];
  #pragma unroll
  for (int k = 0; k < 16; k++)
    cat[k] = wctx[n * 16 + k] + wh[n * 16 + k] + wfg[n * 16 + k] + wedg[n * 16 + k];
  float o4[4];
  #pragma unroll
  for (int r = 0; r < 4; r++) {
    float a = sb4[r];
    #pragma unroll
    for (int k = 0; k < 16; k++) a += sW[r * 16 + k] * cat[k];
    o4[r] = a;
  }
  const float mu0 = o4[0], mu1 = o4[1];
  const float s0 = fminf(4.f, fmaxf(-9.f, o4[2]));
  const float s1 = fminf(4.f, fmaxf(-9.f, o4[3]));
  const float z0 = (target[n * 2 + 0] - mu0) * __expf(-s0);
  const float z1 = (target[n * 2 + 1] - mu1) * __expf(-s1);
  red[tid] = (0.5f * LOG2PI + s0 + 0.5f * z0 * z0) +
             (0.5f * LOG2PI + s1 + 0.5f * z1 * z1);
  __syncthreads();
  for (int off = 256; off > 0; off >>= 1) {
    if (tid < off) red[tid] += red[tid + off];
    __syncthreads();
  }
  if (tid == 0) wnll[0] += red[0] / (float)(NAGT * 2);

  wmu[n * 2 + 0] = mu0;
  wmu[n * 2 + 1] = mu1;
  wcurrs[n * 2 + 0] = wcurr[n * 2 + 0] + mu0;
  wcurrs[n * 2 + 1] = wcurr[n * 2 + 1] + mu1;
  out_sampled[n * 2 + 0] = mu0;
  out_sampled[n * 2 + 1] = mu1;
}

// ---------------------------------------------------------------------------
// Kernel 5 (per step): correction head, carry updates, outputs.
// ---------------------------------------------------------------------------
__global__ void dec_fin_kernel(
    const float* __restrict__ wctx2,
    const float* __restrict__ corr_W, const float* __restrict__ corr_b,
    const float* __restrict__ wmu,
    float* __restrict__ wcurr, float* __restrict__ wout,
    float* __restrict__ out_pred,
    const float* __restrict__ wnll, float* __restrict__ nll_out, int write_nll)
{
  __shared__ float sW[32], sb2[2];
  const int tid = threadIdx.x;
  if (tid < 32) sW[tid] = corr_W[tid];
  if (tid < 2) sb2[tid] = corr_b[tid];
  __syncthreads();

  const int n = tid;
  float a0 = sb2[0], a1 = sb2[1];
  #pragma unroll
  for (int k = 0; k < 16; k++) {
    const float cv = wctx2[n * 16 + k];
    a0 += sW[k] * cv;
    a1 += sW[16 + k] * cv;
  }
  const float op0 = a0 + wmu[n * 2 + 0];
  const float op1 = a1 + wmu[n * 2 + 1];
  out_pred[n * 2 + 0] = op0;
  out_pred[n * 2 + 1] = op1;
  wout[n * 2 + 0] = op0;
  wout[n * 2 + 1] = op1;
  wcurr[n * 2 + 0] += op0;
  wcurr[n * 2 + 1] += op1;
  if (write_nll && tid == 0) nll_out[0] = wnll[0];
}

// ---------------------------------------------------------------------------
// d_in flattening assumption (setup_inputs() insertion order, params nested):
// 0 traj_rel 1 obs_traj_pos 2 nei_index 3 sample_goal 4 h0 5 c0
// 6 enc_W 7 enc_b 8 goal_W 9 goal_b 10 gd_W 11 gd_b 12 dec_W 13 dec_b
// 14 h2p_W 15 h2p_b 16 corr_W 17 corr_b
// 18-21 traj_lstm{Wih,Whh,bih,bhh} 22-25 pred_lstm{...}
// 26-31 pn{se_W,se_b,m1_W,m1_b,m2_W,m2_b} 32-37 pnc{...}
// d_out: pred_rel(12*512*2) | nll(1) | pred_rel_sampled(12*512*2)
// ---------------------------------------------------------------------------
extern "C" void kernel_launch(void* const* d_in, const int* in_sizes, int n_in,
                              void* d_out, int out_size, void* d_ws, size_t ws_size,
                              hipStream_t stream)
{
  (void)in_sizes; (void)n_in; (void)out_size; (void)ws_size;
  const float* traj_rel = (const float*)d_in[0];
  const float* obs_pos  = (const float*)d_in[1];
  const int*   nei      = (const int*)d_in[2];
  const float* goal     = (const float*)d_in[3];
  const float* h0       = (const float*)d_in[4];
  const float* c0       = (const float*)d_in[5];
  const float* enc_W = (const float*)d_in[6],  *enc_b = (const float*)d_in[7];
  const float* goal_W = (const float*)d_in[8], *goal_b = (const float*)d_in[9];
  const float* gd_W = (const float*)d_in[10],  *gd_b = (const float*)d_in[11];
  const float* dec_W = (const float*)d_in[12], *dec_b = (const float*)d_in[13];
  const float* h2p_W = (const float*)d_in[14], *h2p_b = (const float*)d_in[15];
  const float* corr_W = (const float*)d_in[16], *corr_b = (const float*)d_in[17];
  const float* tlWih = (const float*)d_in[18], *tlWhh = (const float*)d_in[19];
  const float* tlbih = (const float*)d_in[20], *tlbhh = (const float*)d_in[21];
  const float* plWih = (const float*)d_in[22], *plWhh = (const float*)d_in[23];
  const float* plbih = (const float*)d_in[24], *plbhh = (const float*)d_in[25];
  const float* pn_seW = (const float*)d_in[26], *pn_seb = (const float*)d_in[27];
  const float* pn_m1W = (const float*)d_in[28], *pn_m1b = (const float*)d_in[29];
  const float* pn_m2W = (const float*)d_in[30], *pn_m2b = (const float*)d_in[31];
  const float* pc_seW = (const float*)d_in[32], *pc_seb = (const float*)d_in[33];
  const float* pc_m1W = (const float*)d_in[34], *pc_m1b = (const float*)d_in[35];
  const float* pc_m2W = (const float*)d_in[36], *pc_m2b = (const float*)d_in[37];

  float* ws = (float*)d_ws;
  float* wh     = ws;                 // 512*16
  float* wc     = wh    + NAGT * 16;  // 512*16
  float* wctx   = wc    + NAGT * 16;  // 512*16
  float* wctx2  = wctx  + NAGT * 16;  // 512*16
  float* wfg    = wctx2 + NAGT * 16;  // 512*16
  float* wedg   = wfg   + NAGT * 16;  // 512*16
  float* wout_  = wedg  + NAGT * 16;  // 512*2
  float* wcurr  = wout_ + NAGT * 2;   // 512*2
  float* wcurrs = wcurr + NAGT * 2;   // 512*2
  float* wmu    = wcurrs + NAGT * 2;  // 512*2
  float* wnll   = wmu   + NAGT * 2;   // 1

  float* pred_rel = (float*)d_out;
  float* nll_out  = pred_rel + PRED_LEN * NAGT * 2;
  float* sampled  = nll_out + 1;

  init_encoder_kernel<<<1, 512, 0, stream>>>(
      traj_rel, obs_pos, goal, h0, c0, enc_W, enc_b, goal_W, goal_b,
      tlWih, tlWhh, tlbih, tlbhh, wh, wc, wctx, wout_, wcurr, wfg, wnll);

  for (int t = 0; t < PRED_LEN; t++) {
    dec_pre_kernel<<<1, 512, 0, stream>>>(
        goal, gd_W, gd_b, dec_W, dec_b, plWih, plWhh, plbih, plbhh,
        wctx, wout_, wcurr, wh, wc, wedg);
    pool_kernel<<<64, 256, 0, stream>>>(
        wcurr, wh, nei + t * NAGT * NAGT,
        pn_seW, pn_seb, pn_m1W, pn_m1b, pn_m2W, pn_m2b, wctx);
    dec_head_kernel<<<1, 512, 0, stream>>>(
        wctx, wh, wfg, wedg, h2p_W, h2p_b,
        traj_rel + (OBS_LEN + t) * NAGT * 2, wcurr,
        wmu, wcurrs, wnll, sampled + t * NAGT * 2);
    pool_kernel<<<64, 256, 0, stream>>>(
        wcurrs, wh, nei + t * NAGT * NAGT,
        pc_seW, pc_seb, pc_m1W, pc_m1b, pc_m2W, pc_m2b, wctx2);
    dec_fin_kernel<<<1, 512, 0, stream>>>(
        wctx2, corr_W, corr_b, wmu, wcurr, wout_,
        pred_rel + t * NAGT * 2, wnll, nll_out, (t == PRED_LEN - 1) ? 1 : 0);
  }
}
